// WorkingMemoryBuffer_34583076667455
// MI455X (gfx1250) — matrix operand order repaired and checked by reading, not compiled, not run on hardware
//
#include <hip/hip_runtime.h>

// ---------------- types for WMMA ----------------
typedef __attribute__((ext_vector_type(16))) __bf16 bf16x16;
typedef __attribute__((ext_vector_type(8)))  unsigned short u16x8;
typedef __attribute__((ext_vector_type(8)))  float  f32x8;

union BFrag { bf16x16 v; u16x8 h[2]; };

// ---------------- problem constants ----------------
#define NB   32          // batch
#define DD   1024        // model dim
#define NN   65536       // buffer rows
#define KK   4096        // top-k
#define HH   8           // heads
#define HD   128         // head dim
#define SCALE 0.08838834764831845f  // 128^-0.5

// ---------------- ws layout (bytes) ----------------
#define OFF_CTRL   ((size_t)0x0)         // [0]=prefix [1]=kk [2]=cntHi [3]=cntEq [4..259]=hist
#define OFF_SEL    ((size_t)0x1000)      // 32 KB: 4096 u64
#define OFF_IDX    ((size_t)0x9000)      // 16 KB: 4096 int
#define OFF_QBF    ((size_t)0xD000)      // 64 KB: query bf16 [32][1024]
#define OFF_CTXB   ((size_t)0x1D000)     // 64 KB: ctx bf16 [32][1024]
#define OFF_QPROJB ((size_t)0x2D000)     // 64 KB: q proj bf16 [32][1024]
#define OFF_CTXF   ((size_t)0x4D000)     // 128 KB: ctx fp32 [32][1024]
#define OFF_WQP    ((size_t)0x80000)     // 2 MB packed bf16 weights
#define OFF_WKP    ((size_t)0x280000)
#define OFF_WVP    ((size_t)0x480000)
#define OFF_WOP    ((size_t)0x680000)
#define OFF_SUB    ((size_t)0x880000)    // 8 MB: subset bf16 [4096][1024]
#define OFF_KPT    ((size_t)0x1080000)   // 8 MB: K^T per-head packed B-fragments
#define OFF_VPT    ((size_t)0x1880000)   // 8 MB: V per-head packed B-fragments
#define OFF_ATTN   ((size_t)0x2080000)   // 4 MB: attn fp32 [32][8][4096]
#define OFF_ATTNB  ((size_t)0x2480000)   // 2 MB: attn bf16 [32][8][4096]

// ---------------- bf16 helpers (integer-only, RNE) ----------------
__device__ __forceinline__ unsigned short f2bf(float f) {
  unsigned int u = __float_as_uint(f);
  unsigned int r = u + 0x7FFFu + ((u >> 16) & 1u);
  return (unsigned short)(r >> 16);
}
__device__ __forceinline__ float bf2f(unsigned short h) {
  return __uint_as_float(((unsigned int)h) << 16);
}
__device__ __forceinline__ unsigned int sortkey(float f) {
  unsigned int u = __float_as_uint(f);
  return (u & 0x80000000u) ? ~u : (u | 0x80000000u);
}

// ======================= top-k: radix select =======================
__global__ void init_ctrl_kernel(unsigned int* ctrl) {
  ctrl[0] = 0u; ctrl[1] = KK; ctrl[2] = 0u; ctrl[3] = 0u;
}
__global__ void zero_hist_kernel(unsigned int* ctrl) {
  ctrl[4 + threadIdx.x] = 0u;
}
__global__ void hist_kernel(const float* __restrict__ pri, unsigned int* ctrl, int shift) {
  unsigned int i = blockIdx.x * 256u + threadIdx.x;
  unsigned int key = sortkey(pri[i]);
  unsigned int mask = (shift == 24) ? 0u : (0xFFFFFFFFu << (shift + 8));
  if ((key & mask) == ctrl[0])
    atomicAdd(&ctrl[4 + ((key >> shift) & 0xFFu)], 1u);
}
__global__ void select_kernel(unsigned int* ctrl, int shift) {
  if (threadIdx.x != 0) return;
  unsigned int kk = ctrl[1];
  unsigned int cum = 0; int b = 255;
  for (; b >= 0; b--) {
    unsigned int c = ctrl[4 + b];
    if (cum + c >= kk) break;
    cum += c;
  }
  if (b < 0) b = 0;
  ctrl[0] |= ((unsigned int)b) << shift;
  ctrl[1] = kk - cum;
}
__global__ void compact_kernel(const float* __restrict__ pri, unsigned int* ctrl,
                               unsigned long long* __restrict__ sel) {
  unsigned int i = blockIdx.x * 256u + threadIdx.x;
  unsigned int key = sortkey(pri[i]);
  unsigned int T = ctrl[0], r = ctrl[1];
  unsigned long long pk = (((unsigned long long)key) << 32) | (unsigned long long)(65535u - i);
  if (key > T) {
    unsigned int p = atomicAdd(&ctrl[2], 1u);
    if (p < KK) sel[p] = pk;
  } else if (key == T) {
    unsigned int e = atomicAdd(&ctrl[3], 1u);
    if (e < r) sel[KK - r + e] = pk;
  }
}
__global__ __launch_bounds__(1024) void sort_kernel(unsigned long long* __restrict__ sel) {
  __shared__ unsigned long long s[KK];
  int t = threadIdx.x;
  for (int i = t; i < KK; i += 1024) s[i] = sel[i];
  __syncthreads();
  for (int k = 2; k <= KK; k <<= 1) {
    for (int j = k >> 1; j > 0; j >>= 1) {
      for (int i = t; i < KK; i += 1024) {
        int p = i ^ j;
        if (p > i) {
          bool up = ((i & k) != 0);           // descending overall
          unsigned long long a = s[i], b = s[p];
          if ((a > b) == up) { s[i] = b; s[p] = a; }
        }
      }
      __syncthreads();
    }
  }
  for (int i = t; i < KK; i += 1024) sel[i] = s[i];
}
__global__ void extract_idx_kernel(const unsigned long long* __restrict__ sel,
                                   int* __restrict__ idx) {
  int s = blockIdx.x * 256 + threadIdx.x;
  idx[s] = 65535 - (int)(sel[s] & 0xFFFFFFFFull);
}

// ======================= conversions / gather / pack =======================
__global__ void gather_bf16_kernel(const float* __restrict__ buf,
                                   const int* __restrict__ idx,
                                   unsigned short* __restrict__ dst) {
  size_t g = (size_t)blockIdx.x * 256 + threadIdx.x;   // < KK*DD
  size_t k = g >> 10, d = g & 1023;
  dst[g] = f2bf(buf[(size_t)idx[k] * DD + d]);
}
__global__ void f32_to_bf16_kernel(const float* __restrict__ src,
                                   unsigned short* __restrict__ dst) {
  size_t g = (size_t)blockIdx.x * 256 + threadIdx.x;
  dst[g] = f2bf(src[g]);
}
// repack fp32 W[1024][1024] into WMMA B-fragment layout:
// t = ((nt*32 + kt)*32 + lane)*16 + h ; k = kt*32 + (h<8 ? 8*(lane>>4)+h : 16+8*(lane>>4)+h-8) ; n = nt*16 + (lane&15)
__global__ void pack_w_kernel(const float* __restrict__ W,
                              unsigned short* __restrict__ P) {
  size_t t = (size_t)blockIdx.x * 256 + threadIdx.x;   // < 1M
  int h  = (int)(t & 15);
  int l  = (int)((t >> 4) & 31);
  int kt = (int)((t >> 9) & 31);
  int nt = (int)(t >> 14);
  int k = kt * 32 + ((h < 8) ? 8 * (l >> 4) + h : 16 + 8 * (l >> 4) + (h - 8));
  int n = nt * 16 + (l & 15);
  P[t] = f2bf(W[(size_t)k * DD + n]);
}

// ======================= WMMA k-loop (register double-buffered) =======================
// A: row-major, kt stride = 32 halves.  B: packed, kt stride = 512 halves.
__device__ __forceinline__ void wmma_kloop(
    const unsigned short* __restrict__ aptr,
    const unsigned short* __restrict__ bbase, size_t ntStride, int l,
    int ktiles, f32x8 acc[4]) {
  BFrag a, b[4];
  a.h[0] = *(const u16x8*)(aptr);
  a.h[1] = *(const u16x8*)(aptr + 16);
#pragma unroll
  for (int jj = 0; jj < 4; jj++) {
    const unsigned short* bp = bbase + jj * ntStride + (size_t)l * 16;
    b[jj].h[0] = *(const u16x8*)bp;
    b[jj].h[1] = *(const u16x8*)(bp + 8);
  }
#pragma unroll 2
  for (int kt = 0; kt < ktiles - 1; kt++) {
    BFrag an, bn[4];
    const unsigned short* ap = aptr + (size_t)(kt + 1) * 32;
    an.h[0] = *(const u16x8*)ap;
    an.h[1] = *(const u16x8*)(ap + 16);
#pragma unroll
    for (int jj = 0; jj < 4; jj++) {
      const unsigned short* bp = bbase + jj * ntStride + (size_t)(kt + 1) * 512 + (size_t)l * 16;
      bn[jj].h[0] = *(const u16x8*)bp;
      bn[jj].h[1] = *(const u16x8*)(bp + 8);
    }
#pragma unroll
    for (int jj = 0; jj < 4; jj++)
      acc[jj] = __builtin_amdgcn_wmma_f32_16x16x32_bf16(
          false, a.v, false, b[jj].v, (short)0, acc[jj], false, false);
    a = an;
#pragma unroll
    for (int jj = 0; jj < 4; jj++) b[jj] = bn[jj];
  }
#pragma unroll
  for (int jj = 0; jj < 4; jj++)
    acc[jj] = __builtin_amdgcn_wmma_f32_16x16x32_bf16(
        false, a.v, false, b[jj].v, (short)0, acc[jj], false, false);
}

// ======================= projection GEMM (4 waves, LDS-staged A via async) ===========
// C[M,1024] = A_bf16[M,1024] @ Wpacked + bias
// mode: 0 -> fp32 row-major, 1 -> bf16 row-major, 2 -> kpT packed, 3 -> vpT packed
__global__ __launch_bounds__(128) void gemm_proj_kernel(
    const unsigned short* __restrict__ A,
    const unsigned short* __restrict__ Bpack,
    const float* __restrict__ bias, int mode,
    float* __restrict__ outF, unsigned short* __restrict__ outB) {
  __shared__ unsigned short sA[16 * DD];     // 32 KB: 16 rows of A
  const int t  = threadIdx.x;
  const int mt = blockIdx.x;
  const int wv = t >> 5;                     // wave id 0..3
  const int l  = t & 31;
  const int lm = l & 15;
  const int hi = l >> 4;

  // ---- async-stage the 16x1024 A tile (32 KB, contiguous in global) into LDS ----
  {
    const char* gbase = (const char*)(A + (size_t)mt * 16 * DD);
    unsigned ldsBase = (unsigned)(size_t)(void*)sA;
#pragma unroll
    for (int i = 0; i < 16; i++) {
      unsigned off = ((unsigned)t + (unsigned)i * 128u) * 16u;   // 16-byte chunks
      unsigned long long g = (unsigned long long)(size_t)(gbase + off);
      unsigned d = ldsBase + off;
      asm volatile("global_load_async_to_lds_b128 %0, %1, off"
                   :: "v"(d), "v"(g) : "memory");
    }
    asm volatile("s_wait_asynccnt 0x0" ::: "memory");
  }
  __syncthreads();

  f32x8 acc[4] = {};
  const unsigned short* aptr = sA + (size_t)lm * DD + hi * 8;
  const int ngG = blockIdx.y * 4 + wv;       // 0..15, group of 4 n-tiles
  const unsigned short* bbase = Bpack + (size_t)(ngG * 4) * 16384;
  wmma_kloop(aptr, bbase, 16384, l, 32, acc);

  const int rowbase = mt * 16 + hi * 8;
#pragma unroll
  for (int jj = 0; jj < 4; jj++) {
    int col = ngG * 64 + jj * 16 + lm;
    float bb = bias[col];
#pragma unroll
    for (int j = 0; j < 8; j++) {
      float v = acc[jj][j] + bb;
      int row = rowbase + j;
      if (mode == 0) {
        outF[(size_t)row * DD + col] = v;
      } else if (mode == 1) {
        outB[(size_t)row * DD + col] = f2bf(v);
      } else if (mode == 2) {
        // K proj -> per-head K^T packed: row = kpos, col = h*128+d
        int h = col >> 7, dd = col & 127;
        int kt = dd >> 5, r = dd & 31;
        int l2 = (row & 15) | ((r & 8) ? 16 : 0);
        int h2 = (r & 7) | ((r >= 16) ? 8 : 0);
        int nt = row >> 4;
        outB[((((size_t)h * 256 + nt) * 4 + kt) << 9) + l2 * 16 + h2] = f2bf(v);
      } else {
        // V proj -> per-head V packed: row = kpos (reduce dim), col = h*128+d
        int h = col >> 7, d = col & 127;
        int kt = row >> 5, r = row & 31;
        int l2 = (d & 15) | ((r & 8) ? 16 : 0);
        int h2 = (r & 7) | ((r >= 16) ? 8 : 0);
        int nt = d >> 4;
        outB[((((size_t)h * 8 + nt) * 128 + kt) << 9) + l2 * 16 + h2] = f2bf(v);
      }
    }
  }
}

// ======================= attention via WMMA =======================
// scores: per head, q_h[32x128] @ K_h^T[128x4096] -> attn fp32 [b][h][k]
__global__ __launch_bounds__(32) void scores_wmma_kernel(
    const unsigned short* __restrict__ qprojb,
    const unsigned short* __restrict__ kpT,
    float* __restrict__ attn) {
  const int mt = blockIdx.x;     // 0..1 (batch tiles)
  const int ng = blockIdx.y;     // 0..63 (64-col groups over K=4096)
  const int h  = blockIdx.z;     // head
  const int l = threadIdx.x, lm = l & 15, hi = l >> 4;
  f32x8 acc[4] = {};
  const unsigned short* aptr = qprojb + (size_t)(mt * 16 + lm) * DD + h * HD + hi * 8;
  const unsigned short* bbase = kpT + (((size_t)h * 256 + ng * 4) * 4) * 512;
  wmma_kloop(aptr, bbase, 4 * 512, l, 4, acc);
  const int rowbase = mt * 16 + hi * 8;
#pragma unroll
  for (int jj = 0; jj < 4; jj++) {
    int col = ng * 64 + jj * 16 + lm;
#pragma unroll
    for (int j = 0; j < 8; j++)
      attn[((size_t)(rowbase + j) * HH + h) * KK + col] = acc[jj][j] * SCALE;
  }
}
// ctx: per head, attn_h[32x4096] @ V_h[4096x128] -> ctx fp32 [b][h*128+d]
__global__ __launch_bounds__(32) void ctx_wmma_kernel(
    const unsigned short* __restrict__ attnb,
    const unsigned short* __restrict__ vpT,
    float* __restrict__ ctxf) {
  const int mt = blockIdx.x;     // 0..1
  const int ng = blockIdx.y;     // 0..1 (64-col groups over HD=128)
  const int h  = blockIdx.z;
  const int l = threadIdx.x, lm = l & 15, hi = l >> 4;
  f32x8 acc[4] = {};
  const unsigned short* aptr = attnb + ((size_t)(mt * 16 + lm) * HH + h) * KK + hi * 8;
  const unsigned short* bbase = vpT + (((size_t)h * 8 + ng * 4) * 128) * 512;
  wmma_kloop(aptr, bbase, (size_t)128 * 512, l, 128, acc);
  const int rowbase = mt * 16 + hi * 8;
#pragma unroll
  for (int jj = 0; jj < 4; jj++) {
    int col = ng * 64 + jj * 16 + lm;
#pragma unroll
    for (int j = 0; j < 8; j++)
      ctxf[(size_t)(rowbase + j) * DD + h * HD + col] = acc[jj][j];
  }
}

// ======================= softmax / head average =======================
__global__ __launch_bounds__(256) void softmax_kernel(float* __restrict__ attn) {
  float* row = attn + (size_t)blockIdx.x * KK;
  __shared__ float red[256];
  int t = threadIdx.x;
  float m = -3.402823466e38f;
  for (int i = t; i < KK; i += 256) m = fmaxf(m, row[i]);
  red[t] = m; __syncthreads();
  for (int s = 128; s > 0; s >>= 1) { if (t < s) red[t] = fmaxf(red[t], red[t + s]); __syncthreads(); }
  m = red[0]; __syncthreads();
  float sum = 0.f;
  for (int i = t; i < KK; i += 256) { float e = __expf(row[i] - m); row[i] = e; sum += e; }
  red[t] = sum; __syncthreads();
  for (int s = 128; s > 0; s >>= 1) { if (t < s) red[t] += red[t + s]; __syncthreads(); }
  float inv = 1.0f / red[0]; __syncthreads();
  for (int i = t; i < KK; i += 256) row[i] *= inv;
}
__global__ void attn_avg_kernel(const float* __restrict__ attn, float* __restrict__ out) {
  size_t g = (size_t)blockIdx.x * 256 + threadIdx.x;  // < 32*4096
  size_t b = g >> 12, k = g & 4095;
  float s = 0.f;
#pragma unroll
  for (int h = 0; h < HH; h++) s += attn[(b * HH + h) * KK + k];
  out[g] = s * 0.125f;
}

// ======================= launcher =======================
extern "C" void kernel_launch(void* const* d_in, const int* in_sizes, int n_in,
                              void* d_out, int out_size, void* d_ws, size_t ws_size,
                              hipStream_t stream) {
  const float* query = (const float*)d_in[0];
  const float* buffer = (const float*)d_in[1];
  const float* pri = (const float*)d_in[2];
  const float* Wq = (const float*)d_in[3];
  const float* bq = (const float*)d_in[4];
  const float* Wk = (const float*)d_in[5];
  const float* bk = (const float*)d_in[6];
  const float* Wv = (const float*)d_in[7];
  const float* bv = (const float*)d_in[8];
  const float* Wo = (const float*)d_in[9];
  const float* bo = (const float*)d_in[10];
  float* out = (float*)d_out;   // [0,32768): retrieved ; [32768,163840): attn_avg

  char* ws = (char*)d_ws;
  unsigned int*       ctrl   = (unsigned int*)(ws + OFF_CTRL);
  unsigned long long* sel    = (unsigned long long*)(ws + OFF_SEL);
  int*                idx    = (int*)(ws + OFF_IDX);
  unsigned short*     qbf    = (unsigned short*)(ws + OFF_QBF);
  unsigned short*     ctxb   = (unsigned short*)(ws + OFF_CTXB);
  unsigned short*     qprojb = (unsigned short*)(ws + OFF_QPROJB);
  float*              ctxf   = (float*)(ws + OFF_CTXF);
  unsigned short*     wqp    = (unsigned short*)(ws + OFF_WQP);
  unsigned short*     wkp    = (unsigned short*)(ws + OFF_WKP);
  unsigned short*     wvp    = (unsigned short*)(ws + OFF_WVP);
  unsigned short*     wop    = (unsigned short*)(ws + OFF_WOP);
  unsigned short*     sub    = (unsigned short*)(ws + OFF_SUB);
  unsigned short*     kpT    = (unsigned short*)(ws + OFF_KPT);
  unsigned short*     vpT    = (unsigned short*)(ws + OFF_VPT);
  float*              attn   = (float*)(ws + OFF_ATTN);
  unsigned short*     attnb  = (unsigned short*)(ws + OFF_ATTNB);

  // ---- top-k radix select + sort ----
  init_ctrl_kernel<<<1, 1, 0, stream>>>(ctrl);
  const int shifts[4] = {24, 16, 8, 0};
  for (int p = 0; p < 4; p++) {
    zero_hist_kernel<<<1, 256, 0, stream>>>(ctrl);
    hist_kernel<<<NN / 256, 256, 0, stream>>>(pri, ctrl, shifts[p]);
    select_kernel<<<1, 1, 0, stream>>>(ctrl, shifts[p]);
  }
  compact_kernel<<<NN / 256, 256, 0, stream>>>(pri, ctrl, sel);
  sort_kernel<<<1, 1024, 0, stream>>>(sel);
  extract_idx_kernel<<<KK / 256, 256, 0, stream>>>(sel, idx);

  // ---- gather + conversions + weight repack ----
  gather_bf16_kernel<<<(KK * DD) / 256, 256, 0, stream>>>(buffer, idx, sub);
  f32_to_bf16_kernel<<<(NB * DD) / 256, 256, 0, stream>>>(query, qbf);
  pack_w_kernel<<<(DD * DD) / 256, 256, 0, stream>>>(Wq, wqp);
  pack_w_kernel<<<(DD * DD) / 256, 256, 0, stream>>>(Wk, wkp);
  pack_w_kernel<<<(DD * DD) / 256, 256, 0, stream>>>(Wv, wvp);
  pack_w_kernel<<<(DD * DD) / 256, 256, 0, stream>>>(Wo, wop);

  // ---- projections (WMMA, async-LDS-staged A) ----
  gemm_proj_kernel<<<dim3(NB / 16, 4), 128, 0, stream>>>(qbf, wqp, bq, 1, nullptr, qprojb);
  gemm_proj_kernel<<<dim3(KK / 16, 4), 128, 0, stream>>>(sub, wkp, bk, 2, nullptr, kpT);
  gemm_proj_kernel<<<dim3(KK / 16, 4), 128, 0, stream>>>(sub, wvp, bv, 3, nullptr, vpT);

  // ---- attention (WMMA scores -> softmax -> WMMA ctx) ----
  scores_wmma_kernel<<<dim3(NB / 16, KK / 64, HH), 32, 0, stream>>>(qprojb, kpT, attn);
  softmax_kernel<<<NB * HH, 256, 0, stream>>>(attn);
  attn_avg_kernel<<<(NB * KK) / 256, 256, 0, stream>>>(attn, out + NB * DD);
  f32_to_bf16_kernel<<<(NB * HH * KK) / 256, 256, 0, stream>>>(attn, attnb);
  ctx_wmma_kernel<<<dim3(NB / 16, HD / 64, HH), 32, 0, stream>>>(attnb, vpT, ctxf);

  // ---- output projection -> d_out ----
  f32_to_bf16_kernel<<<(NB * DD) / 256, 256, 0, stream>>>(ctxf, ctxb);
  gemm_proj_kernel<<<dim3(NB / 16, 4), 128, 0, stream>>>(ctxb, wop, bo, 0, out, nullptr);
}
